// SampleAttention_23046794510348
// MI455X (gfx1250) — matrix operand order repaired and checked by reading, not compiled, not run on hardware
//
#include <hip/hip_runtime.h>
#include <hip/hip_bf16.h>
#include <math.h>

// ---------------------------------------------------------------------------
// CDNA5 (gfx1250) bf16 WMMA implementation of the reference attention stack.
// Wave32; 16x16x32 bf16 WMMA with f32 accumulation everywhere.
// GEMM: 64x64 tile per wave (4x4 WMMA tiles), 128-thread blocks with
// __launch_bounds__(128,1) so the allocator keeps all 16 accumulators +
// 8 fragments in registers (no spills).
// ---------------------------------------------------------------------------

typedef __bf16 bf16_t;
typedef __attribute__((ext_vector_type(16))) __bf16 v16bf;
typedef __attribute__((ext_vector_type(8)))  __bf16 v8bf;
typedef __attribute__((ext_vector_type(8)))  float  v8f;

#define B_DIM   8
#define T_DIM   1024
#define F_IN    512
#define D_ATT   1024
#define H_DIM   16
#define DK      64
#define M_ROWS  (B_DIM * T_DIM)   // 8192

// ---- fragment loaders (layouts per CDNA5 ISA 7.12.2, wave32) --------------
// 32-bit offsets against a uniform base -> SADDR + voffset addressing.
// A (16x32, 16-bit): lane<16 -> row=lane, K {koff..+7, 16+koff..+7}, koff=0;
// lane>=16 -> row=lane-16, koff=8.
static __device__ inline v16bf load_frag_a(const bf16_t* A, unsigned lda,
                                           unsigned row_base, unsigned kk,
                                           unsigned lane) {
  const unsigned row  = row_base + (lane & 15u);
  const unsigned koff = (lane < 16u) ? 0u : 8u;
  const unsigned off  = row * lda + kk + koff;
  v8bf lo = *(const v8bf*)(A + off);        // K = kk+koff .. +7
  v8bf hi = *(const v8bf*)(A + off + 16);   // K = kk+16+koff .. +7
  v16bf a;
#pragma unroll
  for (int i = 0; i < 8; ++i) { a[i] = lo[i]; a[i + 8] = hi[i]; }
  return a;
}

// B (32x16, 16-bit), Bt stored transposed [N][K] row-major:
// col = col_base + lane%16, K {koff..+7, 16+koff..+7}, koff = (lane<16?0:8).
static __device__ inline v16bf load_frag_b(const bf16_t* Bt, unsigned ldb,
                                           unsigned col_base, unsigned kk,
                                           unsigned lane) {
  const unsigned col  = col_base + (lane & 15u);
  const unsigned koff = (lane < 16u) ? 0u : 8u;
  const unsigned off  = col * ldb + kk + koff;
  v8bf lo = *(const v8bf*)(Bt + off);
  v8bf hi = *(const v8bf*)(Bt + off + 16);
  v16bf b;
#pragma unroll
  for (int i = 0; i < 8; ++i) { b[i] = lo[i]; b[i + 8] = hi[i]; }
  return b;
}

// ---- prep kernels ---------------------------------------------------------
__global__ void cvt_f32_bf16(const float* __restrict__ in,
                             bf16_t* __restrict__ out, unsigned n) {
  unsigned i = blockIdx.x * blockDim.x + threadIdx.x;
  const unsigned stride = gridDim.x * blockDim.x;
  for (; i < n; i += stride) out[i] = (bf16_t)in[i];
}

// W [batch][Kd][Nd] fp32  ->  Wt [batch][Nd][Kd] bf16
__global__ void cvt_transpose(const float* __restrict__ W,
                              bf16_t* __restrict__ Wt, unsigned Kd, unsigned Nd) {
  const unsigned base = blockIdx.y * Kd * Nd;
  const unsigned n = Kd * Nd;
  unsigned i = blockIdx.x * blockDim.x + threadIdx.x;
  const unsigned stride = gridDim.x * blockDim.x;
  for (; i < n; i += stride) {
    unsigned k = i / Nd, c = i - k * Nd;
    Wt[base + c * Kd + k] = (bf16_t)W[base + i];
  }
}

// ---- generic bf16 WMMA GEMM: C = [elu](A @ Bt^T + bias) -------------------
// A [M][K] bf16 row-major, Bt [N][K] bf16 row-major (pre-transposed weights),
// bias fp32 [N]; one 64x64 tile per wave (4x4 of 16x16); batch = blockIdx.y.
// vtrans: store C[m][n] at (m/tdim)*N*tdim + n*tdim + (m%tdim)  (V transpose)
__global__ void __launch_bounds__(128, 1)
gemm_bf16_wmma(const bf16_t* __restrict__ A, unsigned lda, unsigned strideA,
               const bf16_t* __restrict__ Bt, unsigned ldb, unsigned strideB,
               const float* __restrict__ bias, unsigned strideBias,
               bf16_t* __restrict__ C, unsigned strideC,
               unsigned M, unsigned N, unsigned K,
               int do_elu, int vtrans, unsigned tdim) {
  const unsigned lane = threadIdx.x & 31u;
  const unsigned wave = threadIdx.x >> 5;
  const unsigned ntn = N >> 6;                       // 64-wide wave tiles
  const unsigned ntm = M >> 6;
  const unsigned tile = blockIdx.x * (blockDim.x >> 5) + wave;
  if (tile >= ntm * ntn) return;
  const unsigned tm = (tile / ntn) * 64u;
  const unsigned tn = (tile - (tile / ntn) * ntn) * 64u;
  const unsigned batch = blockIdx.y;
  const bf16_t* Ab = A + (size_t)batch * strideA;
  const bf16_t* Bb = Bt + (size_t)batch * strideB;
  const float* biasb = bias + (size_t)batch * strideBias;
  bf16_t* Cb = C + (size_t)batch * strideC;

  v8f acc[4][4] = {};
  for (unsigned kk = 0; kk < K; kk += 32) {
    if (kk + 64 < K) {  // global_prefetch_b8 on the next A strip
      __builtin_prefetch((const void*)(Ab + (tm + (lane & 15u)) * lda + kk + 64u), 0, 0);
    }
    v16bf a[4], b[4];
#pragma unroll
    for (int i = 0; i < 4; ++i) a[i] = load_frag_a(Ab, lda, tm + i * 16u, kk, lane);
#pragma unroll
    for (int j = 0; j < 4; ++j) b[j] = load_frag_b(Bb, ldb, tn + j * 16u, kk, lane);
#pragma unroll
    for (int i = 0; i < 4; ++i)
#pragma unroll
      for (int j = 0; j < 4; ++j)
        acc[i][j] = __builtin_amdgcn_wmma_f32_16x16x32_bf16(
            false, a[i], false, b[j], (short)0, acc[i][j], false, false);
  }

#pragma unroll
  for (int i = 0; i < 4; ++i) {
    const unsigned rbase = tm + i * 16u + ((lane >> 4) << 3);
#pragma unroll
    for (int jt = 0; jt < 4; ++jt) {
      const unsigned col = tn + jt * 16u + (lane & 15u);
      const float bve = biasb[col];
#pragma unroll
      for (int j = 0; j < 8; ++j) {
        const unsigned row = rbase + j;
        float v = acc[i][jt][j] + bve;
        if (do_elu) v = (v > 0.0f) ? v : (__expf(v) - 1.0f);
        unsigned cidx;
        if (vtrans) {
          const unsigned bb = row / tdim, tt = row - bb * tdim;
          cidx = bb * N * tdim + col * tdim + tt;
        } else {
          cidx = row * N + col;
        }
        Cb[cidx] = (bf16_t)v;
      }
    }
  }
}

// ---- flash attention per (b, h, 16-row q tile) ----------------------------
// qh/kh: [H][B*T][DK] bf16.  vt: [H][B][DK][T] bf16 (transposed V).
// Output written directly in interleaved concat layout [B][T][DK*H + h].
__global__ void __launch_bounds__(128, 1)
flash_attn_wmma(const bf16_t* __restrict__ qh,
                const bf16_t* __restrict__ kh,
                const bf16_t* __restrict__ vt,
                bf16_t* __restrict__ concat) {
  __shared__ __align__(16) bf16_t pshm[4][16][32];   // per-wave P tile
  const unsigned lane  = threadIdx.x & 31u;
  const unsigned wave  = threadIdx.x >> 5;
  const unsigned h     = blockIdx.y;
  const unsigned b     = blockIdx.z;
  const unsigned qtile = blockIdx.x * 4u + wave;     // 64 q tiles of 16 rows

  const bf16_t* qhb = qh + (size_t)(h * M_ROWS + b * T_DIM) * DK;
  const bf16_t* khb = kh + (size_t)(h * M_ROWS + b * T_DIM) * DK;
  const bf16_t* vtb = vt + (size_t)(h * B_DIM + b) * DK * T_DIM;

  // Q fragments (16 rows x K=64) loaded once.
  const v16bf aq0 = load_frag_a(qhb, DK, qtile * 16u, 0, lane);
  const v16bf aq1 = load_frag_a(qhb, DK, qtile * 16u, 32, lane);

  v8f o0 = {}, o1 = {}, o2 = {}, o3 = {};
  float mrow[8], lrow[8];
#pragma unroll
  for (int j = 0; j < 8; ++j) { mrow[j] = -3.0e38f; lrow[j] = 0.0f; }
  const float scale = 0.125f;   // 1/sqrt(DK)

  for (unsigned s0 = 0; s0 < T_DIM; s0 += 32) {
    // scores for a 16x32 strip: two 16x16 tiles, K=64 -> 2 WMMAs each
    v8f c0 = {}, c1 = {};
    {
      v16bf bk0 = load_frag_b(khb, DK, s0, 0, lane);
      v16bf bk1 = load_frag_b(khb, DK, s0, 32, lane);
      c0 = __builtin_amdgcn_wmma_f32_16x16x32_bf16(false, aq0, false, bk0, (short)0, c0, false, false);
      c0 = __builtin_amdgcn_wmma_f32_16x16x32_bf16(false, aq1, false, bk1, (short)0, c0, false, false);
      v16bf bk2 = load_frag_b(khb, DK, s0 + 16u, 0, lane);
      v16bf bk3 = load_frag_b(khb, DK, s0 + 16u, 32, lane);
      c1 = __builtin_amdgcn_wmma_f32_16x16x32_bf16(false, aq0, false, bk2, (short)0, c1, false, false);
      c1 = __builtin_amdgcn_wmma_f32_16x16x32_bf16(false, aq1, false, bk3, (short)0, c1, false, false);
    }
    // online softmax update; C/D layout: element j -> row j + 8*(lane>=16),
    // col = lane%16 -> row stats reduce across 16 lanes of each half.
    const unsigned prow = (lane >> 4) << 3;
    const unsigned pcol = lane & 15u;
#pragma unroll
    for (int j = 0; j < 8; ++j) {
      float s_a = c0[j] * scale;
      float s_b = c1[j] * scale;
      float tmax = fmaxf(s_a, s_b);
#pragma unroll
      for (int msk = 1; msk <= 8; msk <<= 1) tmax = fmaxf(tmax, __shfl_xor(tmax, msk, 32));
      const float nm = fmaxf(mrow[j], tmax);
      const float alpha = __expf(mrow[j] - nm);
      const float p_a = __expf(s_a - nm);
      const float p_b = __expf(s_b - nm);
      float rs = p_a + p_b;
#pragma unroll
      for (int msk = 1; msk <= 8; msk <<= 1) rs += __shfl_xor(rs, msk, 32);
      lrow[j] = lrow[j] * alpha + rs;
      mrow[j] = nm;
      o0[j] *= alpha; o1[j] *= alpha; o2[j] *= alpha; o3[j] *= alpha;
      pshm[wave][prow + j][pcol]       = (bf16_t)p_a;  // D-layout -> [row][s]
      pshm[wave][prow + j][pcol + 16u] = (bf16_t)p_b;
    }
    __syncthreads();
    // P (16x32) @ V (32x64): A frag from LDS, B frags from transposed V.
    v16bf ap  = load_frag_a(&pshm[wave][0][0], 32, 0, 0, lane);
    v16bf bv0 = load_frag_b(vtb, T_DIM, 0,  s0, lane);
    v16bf bv1 = load_frag_b(vtb, T_DIM, 16, s0, lane);
    v16bf bv2 = load_frag_b(vtb, T_DIM, 32, s0, lane);
    v16bf bv3 = load_frag_b(vtb, T_DIM, 48, s0, lane);
    o0 = __builtin_amdgcn_wmma_f32_16x16x32_bf16(false, ap, false, bv0, (short)0, o0, false, false);
    o1 = __builtin_amdgcn_wmma_f32_16x16x32_bf16(false, ap, false, bv1, (short)0, o1, false, false);
    o2 = __builtin_amdgcn_wmma_f32_16x16x32_bf16(false, ap, false, bv2, (short)0, o2, false, false);
    o3 = __builtin_amdgcn_wmma_f32_16x16x32_bf16(false, ap, false, bv3, (short)0, o3, false, false);
    __syncthreads();
  }

  // epilogue: normalize and scatter into interleaved concat layout
  const unsigned col16 = lane & 15u;
  const unsigned rbase = (lane >> 4) << 3;
#pragma unroll
  for (int j = 0; j < 8; ++j) {
    const unsigned t = qtile * 16u + rbase + j;
    const float inv = 1.0f / lrow[j];
    const unsigned base = (b * T_DIM + t) * D_ATT + h;
    concat[base + (0u  + col16) * H_DIM] = (bf16_t)(o0[j] * inv);
    concat[base + (16u + col16) * H_DIM] = (bf16_t)(o1[j] * inv);
    concat[base + (32u + col16) * H_DIM] = (bf16_t)(o2[j] * inv);
    concat[base + (48u + col16) * H_DIM] = (bf16_t)(o3[j] * inv);
  }
}

// ---- final fc (d_attn -> 1) + softmax over T per batch --------------------
__global__ void logits_softmax(const bf16_t* __restrict__ outb,
                               const float* __restrict__ Wfc,
                               const float* __restrict__ bfc,
                               float* __restrict__ y) {
  __shared__ float sl[T_DIM];
  __shared__ float red[256];
  const unsigned b = blockIdx.x;
  const unsigned tid = threadIdx.x;
  for (unsigned t = tid; t < T_DIM; t += 256) {
    const v8bf* row8 = (const v8bf*)(outb + (size_t)(b * T_DIM + t) * D_ATT);
    float acc = 0.0f;
    for (int c8 = 0; c8 < D_ATT / 8; ++c8) {
      v8bf v = row8[c8];
#pragma unroll
      for (int i = 0; i < 8; ++i) acc += (float)v[i] * Wfc[c8 * 8 + i];
    }
    sl[t] = acc + bfc[0];
  }
  __syncthreads();
  float lm = -3.0e38f;
  for (unsigned t = tid; t < T_DIM; t += 256) lm = fmaxf(lm, sl[t]);
  red[tid] = lm; __syncthreads();
  for (int s = 128; s > 0; s >>= 1) {
    if (tid < (unsigned)s) red[tid] = fmaxf(red[tid], red[tid + s]);
    __syncthreads();
  }
  const float gmax = red[0];
  __syncthreads();
  float ls = 0.0f;
  for (unsigned t = tid; t < T_DIM; t += 256) ls += __expf(sl[t] - gmax);
  red[tid] = ls; __syncthreads();
  for (int s = 128; s > 0; s >>= 1) {
    if (tid < (unsigned)s) red[tid] += red[tid + s];
    __syncthreads();
  }
  const float inv = 1.0f / red[0];
  for (unsigned t = tid; t < T_DIM; t += 256)
    y[b * T_DIM + t] = __expf(sl[t] - gmax) * inv;
}

// ---------------------------------------------------------------------------
extern "C" void kernel_launch(void* const* d_in, const int* in_sizes, int n_in,
                              void* d_out, int out_size, void* d_ws, size_t ws_size,
                              hipStream_t stream) {
  (void)in_sizes; (void)n_in; (void)out_size; (void)ws_size;

  const float* x     = (const float*)d_in[0];
  const float* Wk_in = (const float*)d_in[1];
  const float* bk_in = (const float*)d_in[2];
  const float* Wq_in = (const float*)d_in[3];
  const float* bq_in = (const float*)d_in[4];
  const float* Wv_in = (const float*)d_in[5];
  const float* bv_in = (const float*)d_in[6];
  const float* Wq    = (const float*)d_in[7];
  const float* bq    = (const float*)d_in[8];
  const float* Wk    = (const float*)d_in[9];
  const float* bk    = (const float*)d_in[10];
  const float* Wv    = (const float*)d_in[11];
  const float* bv    = (const float*)d_in[12];
  const float* Wo    = (const float*)d_in[13];
  const float* bo    = (const float*)d_in[14];
  const float* Wfc   = (const float*)d_in[15];
  const float* bfc   = (const float*)d_in[16];
  float* y = (float*)d_out;

  char* ws = (char*)d_ws;
  size_t off = 0;
  auto alloc = [&](size_t bytes) -> char* {
    char* p = ws + off;
    off = off + ((bytes + 255) & ~(size_t)255);
    return p;
  };

  bf16_t* xb     = (bf16_t*)alloc((size_t)M_ROWS * F_IN * 2);
  bf16_t* wkin_t = (bf16_t*)alloc((size_t)D_ATT * F_IN * 2);
  bf16_t* wqin_t = (bf16_t*)alloc((size_t)D_ATT * F_IN * 2);
  bf16_t* wvin_t = (bf16_t*)alloc((size_t)D_ATT * F_IN * 2);
  bf16_t* wq_t   = (bf16_t*)alloc((size_t)H_DIM * DK * D_ATT * 2);
  bf16_t* wk_t   = (bf16_t*)alloc((size_t)H_DIM * DK * D_ATT * 2);
  bf16_t* wv_t   = (bf16_t*)alloc((size_t)H_DIM * DK * D_ATT * 2);
  bf16_t* wo_t   = (bf16_t*)alloc((size_t)D_ATT * D_ATT * 2);
  bf16_t* Kp     = (bf16_t*)alloc((size_t)M_ROWS * D_ATT * 2);
  bf16_t* Qp     = (bf16_t*)alloc((size_t)M_ROWS * D_ATT * 2);
  bf16_t* Vp     = (bf16_t*)alloc((size_t)M_ROWS * D_ATT * 2);
  bf16_t* qhb    = (bf16_t*)alloc((size_t)H_DIM * M_ROWS * DK * 2);
  bf16_t* khb    = (bf16_t*)alloc((size_t)H_DIM * M_ROWS * DK * 2);
  bf16_t* vtb    = (bf16_t*)alloc((size_t)H_DIM * M_ROWS * DK * 2);
  bf16_t* concatb = Kp;  // Kp dead after per-head projections
  bf16_t* outb    = Qp;  // Qp dead after per-head projections

  // --- prep: bf16 conversion + weight transposes ---
  cvt_f32_bf16<<<2048, 256, 0, stream>>>(x, xb, M_ROWS * F_IN);
  cvt_transpose<<<dim3(512, 1), 256, 0, stream>>>(Wk_in, wkin_t, F_IN, D_ATT);
  cvt_transpose<<<dim3(512, 1), 256, 0, stream>>>(Wq_in, wqin_t, F_IN, D_ATT);
  cvt_transpose<<<dim3(512, 1), 256, 0, stream>>>(Wv_in, wvin_t, F_IN, D_ATT);
  cvt_transpose<<<dim3(64, H_DIM), 256, 0, stream>>>(Wq, wq_t, D_ATT, DK);
  cvt_transpose<<<dim3(64, H_DIM), 256, 0, stream>>>(Wk, wk_t, D_ATT, DK);
  cvt_transpose<<<dim3(64, H_DIM), 256, 0, stream>>>(Wv, wv_t, D_ATT, DK);
  cvt_transpose<<<dim3(1024, 1), 256, 0, stream>>>(Wo, wo_t, D_ATT, D_ATT);

  // --- input projections + ELU: [8192,512] @ [512,1024], 64x64 wave tiles ---
  {
    const int wtiles = (M_ROWS / 64) * (D_ATT / 64);  // 2048
    dim3 grid(wtiles / 4, 1);                         // 4 waves / block
    gemm_bf16_wmma<<<grid, 128, 0, stream>>>(xb, F_IN, 0, wkin_t, F_IN, 0,
        bk_in, 0, Kp, 0, M_ROWS, D_ATT, F_IN, 1, 0, T_DIM);
    gemm_bf16_wmma<<<grid, 128, 0, stream>>>(xb, F_IN, 0, wqin_t, F_IN, 0,
        bq_in, 0, Qp, 0, M_ROWS, D_ATT, F_IN, 1, 0, T_DIM);
    gemm_bf16_wmma<<<grid, 128, 0, stream>>>(xb, F_IN, 0, wvin_t, F_IN, 0,
        bv_in, 0, Vp, 0, M_ROWS, D_ATT, F_IN, 1, 0, T_DIM);
  }

  // --- per-head projections: [8192,1024] @ [1024,64] per head ---
  {
    const int wtiles = (M_ROWS / 64) * (DK / 64);  // 128
    dim3 grid(wtiles / 4, H_DIM);
    const unsigned sB = DK * D_ATT;
    const unsigned sC = M_ROWS * DK;
    gemm_bf16_wmma<<<grid, 128, 0, stream>>>(Qp, D_ATT, 0, wq_t, D_ATT, sB,
        bq, DK, qhb, sC, M_ROWS, DK, D_ATT, 0, 0, T_DIM);
    gemm_bf16_wmma<<<grid, 128, 0, stream>>>(Kp, D_ATT, 0, wk_t, D_ATT, sB,
        bk, DK, khb, sC, M_ROWS, DK, D_ATT, 0, 0, T_DIM);
    // V written transposed per (h,b): [DK][T] for contiguous PV B-fragments
    gemm_bf16_wmma<<<grid, 128, 0, stream>>>(Vp, D_ATT, 0, wv_t, D_ATT, sB,
        bv, DK, vtb, sC, M_ROWS, DK, D_ATT, 0, 1, T_DIM);
  }

  // --- flash attention: grid(16 q-tile blocks x4 waves, H, B) ---
  flash_attn_wmma<<<dim3(T_DIM / 16 / 4, H_DIM, B_DIM), 128, 0, stream>>>(
      qhb, khb, vtb, concatb);

  // --- output projection: [8192,1024] @ [1024,1024] ---
  {
    const int wtiles = (M_ROWS / 64) * (D_ATT / 64);  // 2048
    gemm_bf16_wmma<<<dim3(wtiles / 4, 1), 128, 0, stream>>>(concatb, D_ATT, 0,
        wo_t, D_ATT, 0, bo, 0, outb, 0, M_ROWS, D_ATT, D_ATT, 0, 0, T_DIM);
  }

  // --- final fc + softmax over T ---
  logits_softmax<<<B_DIM, 256, 0, stream>>>(outb, Wfc, bfc, y);
}
